// BiMambaBlock_51986284151373
// MI455X (gfx1250) — matrix operand order repaired and checked by reading, not compiled, not run on hardware
//
#include <hip/hip_runtime.h>
#include <hip/hip_bf16.h>

// ---------------------------------------------------------------------------
// BiMamba block for MI455X (gfx1250, wave32).
// GEMMs: LDS-staged block GEMM on V_WMMA_F32_16X16X32_BF16 (bf16 in, f32
// accumulate). ~30 GFLOP vs ~50 MB unique bytes (>600 flop/byte) -> compute
// bound, so the bf16 tensor path is what the chip rewards. Block (256thr/8
// waves) computes 128x128 C; A/B K-chunks of 32 double-buffered in LDS.
// Staging uses the Tensor Data Mover (TENSOR_LOAD_TO_LDS + s_wait_tensorcnt):
// one wave DMAs each 128x32 bf16 tile (2D descriptor, row stride = ld)
// straight into LDS, freeing VGPRs and vector issue slots; manual coalesced
// staging is the fallback when the builtin is unavailable.
// ---------------------------------------------------------------------------

#define DEV __device__ __forceinline__

static constexpr int D_MODEL = 768;
static constexpr int D_STATE = 16;
static constexpr int D_CONV  = 4;
static constexpr int D_INNER = 1536;
static constexpr int DT_RANK = 48;
static constexpr int DT_PAD  = 64;                    // DT_RANK padded to K%32==0
static constexpr int BATCH   = 2;
static constexpr int SEQ     = 1024;
static constexpr int BL      = BATCH * SEQ;           // 2048 rows
static constexpr int PROJ_W  = DT_RANK + 2 * D_STATE; // 80

typedef __attribute__((ext_vector_type(16))) __bf16 v16bf;
typedef __attribute__((ext_vector_type(8)))  __bf16 v8bf;
typedef __attribute__((ext_vector_type(8)))  float  v8f;
typedef unsigned int u32x4 __attribute__((ext_vector_type(4)));
typedef int          i32x8 __attribute__((ext_vector_type(8)));
typedef int          i32x4 __attribute__((ext_vector_type(4)));

#if __has_builtin(__builtin_amdgcn_tensor_load_to_lds) && \
    __has_builtin(__builtin_amdgcn_s_wait_tensorcnt)
#define USE_TDM 1
#else
#define USE_TDM 0
#endif

DEV unsigned short f2bf(float f) {
    union { float f; unsigned u; } v;
    v.f = f;
    unsigned u = v.u;
    unsigned r = (u + 0x7FFFu + ((u >> 16) & 1u)) >> 16;  // RNE
    return (unsigned short)r;
}

DEV float sigmoidf_fast(float x) { return 1.0f / (1.0f + __expf(-x)); }

// Fragment gathers per CDNA5 ISA 7.12.2 (16-bit A 16x32 / B 32x16, wave32):
DEV v16bf lds_a_frag(const unsigned short* base, int row, int hl) {
    const v8bf a0 = *(const v8bf*)(base + row * 32 + hl * 8);
    const v8bf a1 = *(const v8bf*)(base + row * 32 + 16 + hl * 8);
    return __builtin_shufflevector(a0, a1, 0, 1, 2, 3, 4, 5, 6, 7,
                                   8, 9, 10, 11, 12, 13, 14, 15);
}
DEV v16bf lds_b_frag(const unsigned short* base, int row, int hl) {
    const v8bf b0 = *(const v8bf*)(base + row * 32 + hl * 8);
    const v8bf b1 = *(const v8bf*)(base + row * 32 + 16 + hl * 8);
    return __builtin_shufflevector(b0, b1, 0, 1, 2, 3, 4, 5, 6, 7,
                                   8, 9, 10, 11, 12, 13, 14, 15);
}

#if USE_TDM
// Issue one TDM 2D tile load: tile_d0 x tile_d1 elements (2B each) from a
// row-major tensor (row stride stride0 elements) into contiguous LDS.
// D# packing per CDNA5 ISA 08_async_tensor 8.3/8.4: group0 = {count=1,
// lds_addr, global_addr, type=2}; group1 = {data_size=2B, tensor dims/tile
// dims/stride}; groups 2-3 zero (2D tile). 6-arg builtin form (this
// toolchain): (g0, g1, g2, g3, g4, cpol).
DEV void tdm_load_tile_2d(unsigned lds_byte_addr, unsigned long long gaddr,
                          unsigned tensor_d0, unsigned tensor_d1,
                          unsigned tile_d0, unsigned tile_d1,
                          unsigned long long stride0) {
    u32x4 g0;
    g0[0] = 1u;                                            // count=1 (user D#)
    g0[1] = lds_byte_addr;                                 // lds_addr [63:32]
    g0[2] = (unsigned)(gaddr & 0xffffffffull);             // global_addr lo
    g0[3] = (unsigned)((gaddr >> 32) & 0x1ffffffull)       // global_addr hi
            | (2u << 30);                                  // type=2 ("image")
    i32x8 g1;
    g1[0] = (int)(1u << 16);                               // data_size=1 (2B)
    g1[1] = (int)((tensor_d0 & 0xffffu) << 16);            // tensor_dim0 lo16
    g1[2] = (int)((tensor_d0 >> 16) & 0xffffu)             // tensor_dim0 hi16
            | (int)((tensor_d1 & 0xffffu) << 16);          // tensor_dim1 lo16
    g1[3] = (int)((tensor_d1 >> 16) & 0xffffu)             // tensor_dim1 hi16
            | (int)((tile_d0 & 0xffffu) << 16);            // tile_dim0
    g1[4] = (int)(tile_d1 & 0xffffu);                      // tile_dim1 (dim2=0)
    g1[5] = (int)(stride0 & 0xffffffffull);                // dim0_stride lo32
    g1[6] = (int)((stride0 >> 32) & 0xffffull);            // dim0_stride hi16
    g1[7] = 0;
    const i32x4 z4 = {};
    const i32x8 z8 = {};
    __builtin_amdgcn_tensor_load_to_lds(g0, g1, z4, z4, z8, 0);
}
#endif

// ---------------------------------------------------------------------------
// LayerNorm over D_MODEL, emit bf16 normalized activations for the GEMM path.
// ---------------------------------------------------------------------------
__global__ void ln_kernel(const float* __restrict__ x,
                          const float* __restrict__ gamma,
                          const float* __restrict__ beta,
                          unsigned short* __restrict__ x0h) {
    const int row = blockIdx.x;
    const float* xr = x + (size_t)row * D_MODEL;
    __shared__ float rs[256];
    __shared__ float rs2[256];
    float s = 0.f, s2 = 0.f;
    for (int i = threadIdx.x; i < D_MODEL; i += blockDim.x) {
        float v = xr[i];
        s += v;
        s2 += v * v;
    }
    rs[threadIdx.x] = s;
    rs2[threadIdx.x] = s2;
    __syncthreads();
    for (int off = blockDim.x >> 1; off > 0; off >>= 1) {
        if ((int)threadIdx.x < off) {
            rs[threadIdx.x]  += rs[threadIdx.x + off];
            rs2[threadIdx.x] += rs2[threadIdx.x + off];
        }
        __syncthreads();
    }
    const float mu   = rs[0] / (float)D_MODEL;
    const float var  = rs2[0] / (float)D_MODEL - mu * mu;
    const float rstd = rsqrtf(var + 1e-5f);
    for (int i = threadIdx.x; i < D_MODEL; i += blockDim.x) {
        float v = (xr[i] - mu) * rstd * gamma[i] + beta[i];
        x0h[(size_t)row * D_MODEL + i] = f2bf(v);
    }
}

// float -> bf16 elementwise (weight conversion).
__global__ void cvt_bf16_kernel(const float* __restrict__ src,
                                unsigned short* __restrict__ dst, int n) {
    int i = blockIdx.x * blockDim.x + threadIdx.x;
    if (i < n) dst[i] = f2bf(src[i]);
}

// float (rows x cols, stride srcld) -> bf16 (rows x dstld), zero-pad tail.
__global__ void cvt_pad_kernel(const float* __restrict__ src, int srcld,
                               unsigned short* __restrict__ dst, int dstld,
                               int rows, int cols) {
    int i = blockIdx.x * blockDim.x + threadIdx.x;
    if (i >= rows * dstld) return;
    const int r = i / dstld;
    const int c = i % dstld;
    dst[i] = (c < cols) ? f2bf(src[(size_t)r * srcld + c]) : (unsigned short)0;
}

// ---------------------------------------------------------------------------
// LDS-staged WMMA bf16 GEMM:
//   C[M,N](f32) = A[M,K](bf16,row lda) * B[N,K]^T(bf16,row ldb)
// Block (256 thr / 8 waves) computes a 128x128 tile of C, K-chunks of 32.
// Double-buffered LDS staging via TDM (or manual fallback); per chunk each
// wave does 8 WMMAs on a 32x64 sub-block (2 A-frags x 4 B-frags from LDS).
// Requirements: M%128==0, K%32==0; N ragged handled by TDM OOB zero-fill
// (or row-clamp in fallback) + store guard.
// ---------------------------------------------------------------------------
__global__ __launch_bounds__(256)
void gemm_lds_wmma(const unsigned short* __restrict__ A, int lda,
                   const unsigned short* __restrict__ Bw, int ldb,
                   float* __restrict__ C, int ldc,
                   int M, int N, int K, int accumulate) {
    constexpr int BM = 128, BN = 128, BK = 32;
    __shared__ __align__(16) unsigned short lA[2][BM * BK];
    __shared__ __align__(16) unsigned short lB[2][BN * BK];

    const int tid  = threadIdx.x;
    const int lane = tid & 31;
    const int w    = tid >> 5;      // wave 0..7
    const int wm   = w >> 1;        // 0..3 : 32-row strip
    const int wn   = w & 1;         // 0..1 : 64-col strip
    const int hl   = lane >> 4;     // half-wave
    const int l15  = lane & 15;

    const int nBlocksN = (N + BN - 1) / BN;
    const int bm = blockIdx.x / nBlocksN;
    const int bn = blockIdx.x % nBlocksN;

    v8f acc[2][4];
#pragma unroll
    for (int mi = 0; mi < 2; ++mi)
#pragma unroll
        for (int q = 0; q < 4; ++q) acc[mi][q] = (v8f){};

    const int nk = K / BK;

#if USE_TDM
    // TDM staging: wave 0 issues one 2D tile DMA per operand per chunk.
    const unsigned long long gAbase =
        (unsigned long long)(size_t)(A  + (size_t)(bm * BM) * lda);
    const unsigned long long gBbase =
        (unsigned long long)(size_t)(Bw + (size_t)(bn * BN) * ldb);
    const unsigned aRows = (unsigned)(M - bm * BM < BM ? M - bm * BM : BM);
    const unsigned bRows = (unsigned)(N - bn * BN < BN ? N - bn * BN : BN);

    if (w == 0) {
        tdm_load_tile_2d((unsigned)(size_t)&lA[0][0], gAbase,
                         (unsigned)K, aRows, BK, BM, (unsigned long long)lda);
        tdm_load_tile_2d((unsigned)(size_t)&lB[0][0], gBbase,
                         (unsigned)K, bRows, BK, BN, (unsigned long long)ldb);
    }
    __builtin_amdgcn_s_wait_tensorcnt(0);
    __syncthreads();
#else
    // manual staging: thread -> (row, 16-half column chunk)
    const int srow = tid >> 1;          // 0..127
    const int scol = (tid & 1) * 16;    // 0 | 16
    int garow = bm * BM + srow; if (garow >= M) garow = M - 1;
    int gbrow = bn * BN + srow; if (gbrow >= N) gbrow = N - 1;
    const unsigned short* gA = A  + (size_t)garow * lda + scol;
    const unsigned short* gB = Bw + (size_t)gbrow * ldb + scol;
    const int soff = srow * BK + scol;
    {
        const uint4 a0 = *(const uint4*)(gA);
        const uint4 a1 = *(const uint4*)(gA + 8);
        const uint4 b0 = *(const uint4*)(gB);
        const uint4 b1 = *(const uint4*)(gB + 8);
        *(uint4*)(&lA[0][soff])     = a0;
        *(uint4*)(&lA[0][soff] + 8) = a1;
        *(uint4*)(&lB[0][soff])     = b0;
        *(uint4*)(&lB[0][soff] + 8) = b1;
    }
    __syncthreads();
#endif

    for (int k = 0; k < nk; ++k) {
        const int p = k & 1;
        const bool more = (k + 1 < nk);

#if USE_TDM
        if (more && w == 0) {
            const int pn = p ^ 1;
            const unsigned long long koff =
                (unsigned long long)(k + 1) * BK * 2ull;
            tdm_load_tile_2d((unsigned)(size_t)&lA[pn][0], gAbase + koff,
                             (unsigned)K, aRows, BK, BM,
                             (unsigned long long)lda);
            tdm_load_tile_2d((unsigned)(size_t)&lB[pn][0], gBbase + koff,
                             (unsigned)K, bRows, BK, BN,
                             (unsigned long long)ldb);
        }
#else
        uint4 a0 = {}, a1 = {}, b0 = {}, b1 = {};
        if (more) {
            const int k0 = (k + 1) * BK;
            a0 = *(const uint4*)(gA + k0);
            a1 = *(const uint4*)(gA + k0 + 8);
            b0 = *(const uint4*)(gB + k0);
            b1 = *(const uint4*)(gB + k0 + 8);
        }
#endif

        // compute chunk k from LDS[p] (staging for k+1 in flight above)
        const v16bf af0 = lds_a_frag(&lA[p][0], wm * 32 + l15, hl);
        const v16bf af1 = lds_a_frag(&lA[p][0], wm * 32 + 16 + l15, hl);
        v16bf bf[4];
#pragma unroll
        for (int q = 0; q < 4; ++q)
            bf[q] = lds_b_frag(&lB[p][0], wn * 64 + q * 16 + l15, hl);
#pragma unroll
        for (int q = 0; q < 4; ++q) {
            acc[0][q] = __builtin_amdgcn_wmma_f32_16x16x32_bf16(
                false, af0, false, bf[q], (short)0, acc[0][q], false, false);
            acc[1][q] = __builtin_amdgcn_wmma_f32_16x16x32_bf16(
                false, af1, false, bf[q], (short)0, acc[1][q], false, false);
        }

#if USE_TDM
        __builtin_amdgcn_s_wait_tensorcnt(0);
#else
        if (more) {
            const int pn = p ^ 1;
            *(uint4*)(&lA[pn][soff])     = a0;
            *(uint4*)(&lA[pn][soff] + 8) = a1;
            *(uint4*)(&lB[pn][soff])     = b0;
            *(uint4*)(&lB[pn][soff] + 8) = b1;
        }
#endif
        __syncthreads();
    }

    // store 32x64 per wave
#pragma unroll
    for (int mi = 0; mi < 2; ++mi) {
#pragma unroll
        for (int q = 0; q < 4; ++q) {
            const int n = bn * BN + wn * 64 + q * 16 + l15;
#pragma unroll
            for (int r = 0; r < 8; ++r) {
                const int m = bm * BM + wm * 32 + mi * 16 + hl * 8 + r;
                if (m < M && n < N) {
                    float v = acc[mi][q][r];
                    if (accumulate) v += C[(size_t)m * ldc + n];
                    C[(size_t)m * ldc + n] = v;
                }
            }
        }
    }
}

// ---------------------------------------------------------------------------
// Causal depthwise conv (K=4) + SiLU. dir=0: forward (left pad);
// dir=1: time-reversed conv (== flip->conv->flip).
// ---------------------------------------------------------------------------
__global__ void conv_silu_kernel(const float* __restrict__ xz,
                                 const float* __restrict__ cw,
                                 const float* __restrict__ cb,
                                 float* __restrict__ xc,
                                 unsigned short* __restrict__ xch,
                                 int dir) {
    int i = blockIdx.x * blockDim.x + threadIdx.x;
    if (i >= BL * D_INNER) return;
    const int d   = i % D_INNER;
    const int row = i / D_INNER;
    const int t   = row % SEQ;
    const int b   = row / SEQ;
    float acc = cb[d];
#pragma unroll
    for (int k = 0; k < D_CONV; ++k) {
        const int tt = dir ? (t + (D_CONV - 1) - k) : (t - (D_CONV - 1) + k);
        if (tt >= 0 && tt < SEQ) {
            acc += cw[d * D_CONV + k] *
                   xz[((size_t)(b * SEQ + tt)) * (2 * D_INNER) + d];
        }
    }
    const float sv = acc * sigmoidf_fast(acc);
    xc[i] = sv;
    xch[i] = f2bf(sv);
}

// proj[:, 0:DT_RANK] -> bf16 (BL x DT_PAD), zero-padded, for the dt GEMM.
__global__ void dt_cvt_kernel(const float* __restrict__ proj,
                              unsigned short* __restrict__ dth) {
    int i = blockIdx.x * blockDim.x + threadIdx.x;
    if (i >= BL * DT_PAD) return;
    const int row = i / DT_PAD;
    const int j   = i % DT_PAD;
    dth[i] = (j < DT_RANK) ? f2bf(proj[(size_t)row * PROJ_W + j])
                           : (unsigned short)0;
}

// delta = softplus(delta_pre + dt_b)
__global__ void softplus_kernel(float* __restrict__ delta,
                                const float* __restrict__ dtb) {
    int i = blockIdx.x * blockDim.x + threadIdx.x;
    if (i >= BL * D_INNER) return;
    const int d = i % D_INNER;
    const float v = delta[i] + dtb[d];
    delta[i] = (v > 20.f) ? v : log1pf(__expf(v));
}

// ---------------------------------------------------------------------------
// Selective scan: one thread per (batch, channel), 16 states in registers,
// sequential over SEQ (dir selects order). Fuses +xc*D and SiLU(z) gate,
// emits bf16 y for the out_proj WMMA GEMM.
// ---------------------------------------------------------------------------
__global__ void scan_kernel(const float* __restrict__ delta,
                            const float* __restrict__ xc,
                            const float* __restrict__ proj,
                            const float* __restrict__ xz,
                            const float* __restrict__ A_log,
                            const float* __restrict__ Dp,
                            unsigned short* __restrict__ yh,
                            int dir) {
    int tid = blockIdx.x * blockDim.x + threadIdx.x;
    if (tid >= BATCH * D_INNER) return;
    const int d = tid % D_INNER;
    const int b = tid / D_INNER;

    float A[D_STATE];
#pragma unroll
    for (int n = 0; n < D_STATE; ++n)
        A[n] = -__expf(A_log[(size_t)d * D_STATE + n]);
    const float Dd = Dp[d];

    float h[D_STATE];
#pragma unroll
    for (int n = 0; n < D_STATE; ++n) h[n] = 0.f;

    for (int s = 0; s < SEQ; ++s) {
        const int t = dir ? (SEQ - 1 - s) : s;
        const size_t row = (size_t)b * SEQ + t;
        const float dl = delta[row * D_INNER + d];
        const float xv = xc[row * D_INNER + d];
        const float dx = dl * xv;
        const float* pr = proj + row * PROJ_W;
        float y = 0.f;
#pragma unroll
        for (int n = 0; n < D_STATE; ++n) {
            const float dA = __expf(dl * A[n]);
            h[n] = dA * h[n] + dx * pr[DT_RANK + n];      // + dx * B_t[n]
            y += h[n] * pr[DT_RANK + D_STATE + n];        // * C_t[n]
        }
        y += xv * Dd;
        const float z = xz[row * (2 * D_INNER) + D_INNER + d];
        y *= z * sigmoidf_fast(z);
        yh[row * D_INNER + d] = f2bf(y);
    }
}

// out = x + (out_fwd + out_bwd)
__global__ void residual_kernel(const float* __restrict__ x,
                                const float* __restrict__ acc,
                                float* __restrict__ out, int n) {
    int i = blockIdx.x * blockDim.x + threadIdx.x;
    if (i < n) out[i] = x[i] + acc[i];
}

// ---------------------------------------------------------------------------
// Host-side orchestration.
// ---------------------------------------------------------------------------
extern "C" void kernel_launch(void* const* d_in, const int* in_sizes, int n_in,
                              void* d_out, int out_size, void* d_ws, size_t ws_size,
                              hipStream_t stream) {
    (void)in_sizes; (void)n_in; (void)out_size; (void)ws_size;

    const float* x     = (const float*)d_in[0];
    const float* gamma = (const float*)d_in[1];
    const float* beta  = (const float*)d_in[2];

    // bump allocator on d_ws (256B aligned -> all GEMM rows 16B+ aligned)
    char* base = (char*)d_ws;
    size_t off = 0;
    auto alloc = [&](size_t bytes) -> void* {
        off = (off + 255) & ~(size_t)255;
        void* p = base + off;
        off += bytes;
        return p;
    };

    unsigned short* x0h  = (unsigned short*)alloc((size_t)BL * D_MODEL * 2);
    unsigned short* inwh = (unsigned short*)alloc((size_t)2 * D_INNER * D_MODEL * 2);
    unsigned short* xpwh = (unsigned short*)alloc((size_t)PROJ_W * D_INNER * 2);
    unsigned short* dtwh = (unsigned short*)alloc((size_t)D_INNER * DT_PAD * 2);
    unsigned short* owh  = (unsigned short*)alloc((size_t)D_MODEL * D_INNER * 2);
    float* xz            = (float*)alloc((size_t)BL * 2 * D_INNER * 4);
    float* xc            = (float*)alloc((size_t)BL * D_INNER * 4);
    unsigned short* xch  = (unsigned short*)alloc((size_t)BL * D_INNER * 2);
    float* proj          = (float*)alloc((size_t)BL * PROJ_W * 4);
    unsigned short* dth  = (unsigned short*)alloc((size_t)BL * DT_PAD * 2);
    float* delta         = (float*)alloc((size_t)BL * D_INNER * 4);
    unsigned short* yh   = (unsigned short*)alloc((size_t)BL * D_INNER * 2);
    float* oacc          = (float*)alloc((size_t)BL * D_MODEL * 4);

    auto cvt = [&](const float* src, unsigned short* dst, int n) {
        cvt_bf16_kernel<<<(n + 255) / 256, 256, 0, stream>>>(src, dst, n);
    };
    auto gemm = [&](const unsigned short* A, int lda,
                    const unsigned short* B, int ldb,
                    float* C, int ldc, int M, int N, int K, int accFlag) {
        const int blocks = (M / 128) * ((N + 127) / 128);
        gemm_lds_wmma<<<blocks, 256, 0, stream>>>(A, lda, B, ldb, C, ldc,
                                                  M, N, K, accFlag);
    };

    // LayerNorm (shared by both directions)
    ln_kernel<<<BL, 256, 0, stream>>>(x, gamma, beta, x0h);

    for (int dir = 0; dir < 2; ++dir) {
        const int p = 3 + dir * 9;
        const float* in_w    = (const float*)d_in[p + 0];
        const float* conv_w  = (const float*)d_in[p + 1];
        const float* conv_b  = (const float*)d_in[p + 2];
        const float* xproj_w = (const float*)d_in[p + 3];
        const float* dt_w    = (const float*)d_in[p + 4];
        const float* dt_b    = (const float*)d_in[p + 5];
        const float* A_log   = (const float*)d_in[p + 6];
        const float* Dp      = (const float*)d_in[p + 7];
        const float* out_w   = (const float*)d_in[p + 8];

        // weights -> bf16 (dt_w padded 48 -> 64 cols)
        cvt(in_w,    inwh, 2 * D_INNER * D_MODEL);
        cvt(xproj_w, xpwh, PROJ_W * D_INNER);
        cvt(out_w,   owh,  D_MODEL * D_INNER);
        {
            const int n = D_INNER * DT_PAD;
            cvt_pad_kernel<<<(n + 255) / 256, 256, 0, stream>>>(
                dt_w, DT_RANK, dtwh, DT_PAD, D_INNER, DT_RANK);
        }

        // xz = x0 @ in_w.T : (2048 x 768) * (3072 x 768)^T
        gemm(x0h, D_MODEL, inwh, D_MODEL, xz, 2 * D_INNER,
             BL, 2 * D_INNER, D_MODEL, 0);

        // causal depthwise conv + SiLU (direction-aware)
        {
            const int n = BL * D_INNER;
            conv_silu_kernel<<<(n + 255) / 256, 256, 0, stream>>>(
                xz, conv_w, conv_b, xc, xch, dir);
        }

        // proj = xc @ xproj_w.T : (2048 x 1536) * (80 x 1536)^T
        gemm(xch, D_INNER, xpwh, D_INNER, proj, PROJ_W,
             BL, PROJ_W, D_INNER, 0);

        // dt part -> bf16 padded (BL x 64)
        {
            const int n = BL * DT_PAD;
            dt_cvt_kernel<<<(n + 255) / 256, 256, 0, stream>>>(proj, dth);
        }

        // delta_pre = dt @ dt_w.T : (2048 x 64pad) * (1536 x 64pad)^T
        gemm(dth, DT_PAD, dtwh, DT_PAD, delta, D_INNER,
             BL, D_INNER, DT_PAD, 0);

        // delta = softplus(delta_pre + dt_b)
        {
            const int n = BL * D_INNER;
            softplus_kernel<<<(n + 255) / 256, 256, 0, stream>>>(delta, dt_b);
        }

        // selective scan + D skip + SiLU(z) gate -> bf16 y
        {
            const int n = BATCH * D_INNER;
            scan_kernel<<<(n + 255) / 256, 256, 0, stream>>>(
                delta, xc, proj, xz, A_log, Dp, yh, dir);
        }

        // out_dir = y @ out_w.T : (2048 x 1536) * (768 x 1536)^T ; dir 1 accumulates
        gemm(yh, D_INNER, owh, D_INNER, oacc, D_MODEL,
             BL, D_MODEL, D_INNER, dir);
    }

    // out = x + out_fwd + out_bwd
    {
        const int n = BL * D_MODEL;
        residual_kernel<<<(n + 255) / 256, 256, 0, stream>>>(x, oacc,
                                                             (float*)d_out, n);
    }
}
